// DifferentialMultiHeadAttention_33320356282620
// MI455X (gfx1250) — matrix operand order repaired and checked by reading, not compiled, not run on hardware
//
#include <hip/hip_runtime.h>

// ---------------------------------------------------------------------------
// Problem constants (match reference)
// ---------------------------------------------------------------------------
static constexpr int BB   = 16;
static constexpr int SS   = 577;
static constexpr int DD   = 768;
static constexpr int HH   = 12;
static constexpr int DHD  = 64;             // head dim
static constexpr int MS   = BB * SS;        // 9232 rows = 577 * 16 (tile exact)
static constexpr int SP   = 608;            // S padded to multiple of 32 (19*32)
static constexpr int CPB  = DD * SS;        // 443136 floats per batch in c-buffer

typedef __attribute__((ext_vector_type(16))) __bf16 v16bf;
typedef __attribute__((ext_vector_type(8)))  float  v8f;

typedef unsigned int u32;
typedef u32 u32x4 __attribute__((ext_vector_type(4)));
typedef u32x4 u32x4a __attribute__((may_alias));

// ---------------------------------------------------------------------------
// WMMA helpers (gfx1250: V_WMMA_F32_16X16X32_BF16, wave32)
// ---------------------------------------------------------------------------
__device__ __forceinline__ v8f wmma_bf16(v16bf a, v16bf b, v8f c) {
  return __builtin_amdgcn_wmma_f32_16x16x32_bf16(false, a, false, b,
                                                 (short)0, c, false, false);
}

__device__ __forceinline__ v8f zero_v8f() {
  v8f z;
#pragma unroll
  for (int j = 0; j < 8; ++j) z[j] = 0.0f;
  return z;
}

union frag_u { u32x4a u[2]; v16bf v; };

// A fragment (16-bit A matrix 16x32), lane = 16*hf + r:
//   a[0..7]  = A[r][k0 + hf*8 + j]   (16B contiguous)
//   a[8..15] = A[r][k0 + 16 + hf*8 + j]
// base = A_row + k0 + hf*8 ; 16B aligned.  Works for global and LDS.
__device__ __forceinline__ v16bf load_a_frag(const __bf16* base) {
  frag_u f;
  f.u[0] = *(const u32x4a*)(base);
  f.u[1] = *(const u32x4a*)(base + 16);
  return f.v;
}

// B fragment (16-bit B matrix 32x16), lane = 16*hf + c:
//   b[0..7] = B[k0 + hf*8 + e][c], b[8..15] = B[k0 + 16 + hf*8 + e][c], K contiguous in memory.
__device__ __forceinline__ v16bf load_b_frag(const __bf16* base) {
  frag_u f;
  f.u[0] = *(const u32x4a*)(base);
  f.u[1] = *(const u32x4a*)(base + 16);
  return f.v;
}

// ---------------------------------------------------------------------------
// Utility kernels
// ---------------------------------------------------------------------------
__global__ void k_zero(u32* __restrict__ p, int n) {
  int i = blockIdx.x * 256 + threadIdx.x;
  if (i < n) p[i] = 0u;
}

__global__ void k_cvt(const float* __restrict__ src, __bf16* __restrict__ dst, int n) {
  int i = blockIdx.x * 256 + threadIdx.x;
  if (i < n) dst[i] = (__bf16)src[i];
}

// f32 [K][N] row-major  ->  bf16 [N][K] (transposed, K contiguous)
__global__ void k_cvt_t(const float* __restrict__ src, __bf16* __restrict__ dst,
                        int K, int N) {
  int i = blockIdx.x * 256 + threadIdx.x;
  if (i < K * N) {
    const int k = i / N, n = i - k * N;
    dst[(size_t)n * K + k] = (__bf16)src[i];
  }
}

// ---------------------------------------------------------------------------
// GEMM: C[M,N] = A[M,K](bf16 rowmajor) * Bt[N,K](bf16, pre-transposed) + bias
// One wave per 16x128 tile: 8 accumulators, A fragment reused x8 per K-step.
// Software-pipelined: next K-step's 18 b128 loads issued before current WMMAs.
// ---------------------------------------------------------------------------
__global__ __launch_bounds__(32)
void k_gemm_bf16(const __bf16* __restrict__ A, const __bf16* __restrict__ Bt,
                 const float* __restrict__ bias,
                 float* __restrict__ Cf, __bf16* __restrict__ Cb,
                 __bf16* __restrict__ CtV,
                 int M, int K, int N) {
  const int lane = threadIdx.x & 31;
  const int hf   = lane >> 4;
  const int r    = lane & 15;
  const int m0   = blockIdx.x * 16;
  const int n0   = blockIdx.y * 128;

  v8f acc[8];
#pragma unroll
  for (int t = 0; t < 8; ++t) acc[t] = zero_v8f();

  const __bf16* Arow  = A + (size_t)(m0 + r) * K + hf * 8;
  const __bf16* Bbase = Bt + (size_t)(n0 + r) * K + hf * 8;

  // prologue: load k0 = 0 fragments
  v16bf a_cur = load_a_frag(Arow);
  v16bf b_cur[8];
#pragma unroll
  for (int t = 0; t < 8; ++t) b_cur[t] = load_b_frag(Bbase + (size_t)(t * 16) * K);

#pragma unroll 2
  for (int k0 = 0; k0 < K; k0 += 32) {
    // prefetch next K-step (clamped on the last iteration: redundant reload)
    const int k1 = (k0 + 32 < K) ? (k0 + 32) : k0;
    v16bf a_nxt = load_a_frag(Arow + k1);
    v16bf b_nxt[8];
#pragma unroll
    for (int t = 0; t < 8; ++t)
      b_nxt[t] = load_b_frag(Bbase + (size_t)(t * 16) * K + k1);

#pragma unroll
    for (int t = 0; t < 8; ++t) acc[t] = wmma_bf16(a_cur, b_cur[t], acc[t]);

    a_cur = a_nxt;
#pragma unroll
    for (int t = 0; t < 8; ++t) b_cur[t] = b_nxt[t];
  }

#pragma unroll
  for (int t = 0; t < 8; ++t) {
    const int col = n0 + t * 16 + r;
    const float bv = bias ? bias[col] : 0.0f;
#pragma unroll
    for (int j = 0; j < 8; ++j) {
      const int row = m0 + j + hf * 8;           // D layout: M = j + 8*half
      const float v = acc[t][j] + bv;
      if (Cf) Cf[(size_t)row * N + col] = v;
      if (Cb) Cb[(size_t)row * N + col] = (__bf16)v;
      if (CtV) {
        const int b = row / SS, s = row - b * SS;
        CtV[((size_t)col * BB + b) * SP + s] = (__bf16)v;
      }
    }
  }
}

// ---------------------------------------------------------------------------
// Fused differential attention for one (b, h, 16-query tile).
// vt layout: [(h*64+dh)*BB + b]*SP + s  (K contiguous, tail pre-zeroed).
// Output written directly in GroupNorm layout c[b][(h*64+dh)*S + s].
// ---------------------------------------------------------------------------
__global__ __launch_bounds__(32)
void k_attn(const __bf16* __restrict__ qc, const __bf16* __restrict__ kc,
            const __bf16* __restrict__ vt, const float* __restrict__ lam_p,
            float* __restrict__ cbuf) {
  __shared__ __bf16 sc1[16 * SP];   // scores1; later aliased as diff-P matrix
  __shared__ __bf16 sc2[16 * SP];   // scores2
  __shared__ float  rmax1[16], rinv1[16], rmax2[16], rinv2[16];

  const float lam  = lam_p[0];
  const int  lane  = threadIdx.x;
  const int  hf    = lane >> 4;
  const int  r     = lane & 15;
  const int  bh    = blockIdx.y;
  const int  b     = bh / HH;
  const int  h     = bh % HH;
  const int  q0    = blockIdx.x * 16;
  const float scale = 0.125f;       // 1/sqrt(64)

  // ---- Q fragments, held in VGPRs for the whole key loop -------------------
  const int qs = (q0 + r < SS) ? (q0 + r) : (SS - 1);
  const __bf16* q1p = qc + ((size_t)(b * SS + qs)) * (2 * DD) + h * DHD;
  const __bf16* q2p = q1p + DD;
  const v16bf a10 = load_a_frag(q1p + hf * 8);
  const v16bf a11 = load_a_frag(q1p + 32 + hf * 8);
  const v16bf a20 = load_a_frag(q2p + hf * 8);
  const v16bf a21 = load_a_frag(q2p + 32 + hf * 8);

  // per-key-tile fragment base pointer (column = r, clamped row)
  auto kptr = [&](int kt) -> const __bf16* {
    const int kcol = kt * 16 + r;
    const int ksr  = (kcol < SS) ? kcol : (SS - 1);
    return kc + ((size_t)(b * SS + ksr)) * (2 * DD) + h * DHD + hf * 8;
  };

  // ---- score tiles: 4 WMMAs per 16-key tile, pipelined ---------------------
  {
    const __bf16* kp = kptr(0);
    v16bf b10 = load_b_frag(kp);
    v16bf b11 = load_b_frag(kp + 32);
    v16bf b20 = load_b_frag(kp + DD);
    v16bf b21 = load_b_frag(kp + DD + 32);
    for (int kt = 0; kt < 37; ++kt) {
      const __bf16* kn = kptr((kt + 1 < 37) ? (kt + 1) : kt);
      const v16bf n10 = load_b_frag(kn);
      const v16bf n11 = load_b_frag(kn + 32);
      const v16bf n20 = load_b_frag(kn + DD);
      const v16bf n21 = load_b_frag(kn + DD + 32);

      v8f acc1 = zero_v8f(), acc2 = zero_v8f();
      acc1 = wmma_bf16(a10, b10, acc1);
      acc1 = wmma_bf16(a11, b11, acc1);
      acc2 = wmma_bf16(a20, b20, acc2);
      acc2 = wmma_bf16(a21, b21, acc2);
#pragma unroll
      for (int j = 0; j < 8; ++j) {
        const int row = j + hf * 8;
        sc1[row * SP + kt * 16 + r] = (__bf16)(acc1[j] * scale);
        sc2[row * SP + kt * 16 + r] = (__bf16)(acc2[j] * scale);
      }
      b10 = n10; b11 = n11; b20 = n20; b21 = n21;
    }
  }
  __syncthreads();

  // ---- mask padded key columns [S, SP) -------------------------------------
  for (int i = lane; i < 16 * (SP - SS); i += 32) {
    const int row = i / (SP - SS);
    const int col = SS + (i % (SP - SS));
    sc1[row * SP + col] = (__bf16)(-1e30f);
    sc2[row * SP + col] = (__bf16)(-1e30f);
  }
  __syncthreads();

  // ---- dual softmax stats: 2 lanes per row ---------------------------------
  {
    const int row  = lane >> 1;
    const int part = lane & 1;
    const int c0 = part * (SP / 2), c1 = c0 + SP / 2;
    float m1 = -1e30f, m2 = -1e30f;
    for (int c = c0; c < c1; ++c) {
      m1 = fmaxf(m1, (float)sc1[row * SP + c]);
      m2 = fmaxf(m2, (float)sc2[row * SP + c]);
    }
    m1 = fmaxf(m1, __shfl_xor(m1, 1, 32));
    m2 = fmaxf(m2, __shfl_xor(m2, 1, 32));
    float s1 = 0.0f, s2 = 0.0f;
    for (int c = c0; c < c1; ++c) {
      s1 += __expf((float)sc1[row * SP + c] - m1);
      s2 += __expf((float)sc2[row * SP + c] - m2);
    }
    s1 += __shfl_xor(s1, 1, 32);
    s2 += __shfl_xor(s2, 1, 32);
    if (part == 0) {
      rmax1[row] = m1; rinv1[row] = 1.0f / s1;
      rmax2[row] = m2; rinv2[row] = 1.0f / s2;
    }
  }
  __syncthreads();

  // ---- diff = p1 - lam*p2, written in place over sc1 (elementwise alias) ---
  for (int i = lane; i < 16 * SP; i += 32) {
    const int row = i / SP;
    const float p1 = __expf((float)sc1[i] - rmax1[row]) * rinv1[row];
    const float p2 = __expf((float)sc2[i] - rmax2[row]) * rinv2[row];
    sc1[i] = (__bf16)(p1 - lam * p2);
  }
  __syncthreads();

  // ---- ctx = diff[16 x SP] @ V[SP x 64]; A from LDS, pipelined -------------
  v8f cacc[4];
#pragma unroll
  for (int t = 0; t < 4; ++t) cacc[t] = zero_v8f();

  const __bf16* vbase = vt + ((size_t)(h * DHD + r) * BB + b) * SP + hf * 8;
  const size_t  vstep = (size_t)16 * BB * SP;      // +16 output channels

  v16bf pa = load_a_frag(&sc1[r * SP + hf * 8]);
  v16bf vb_cur[4];
#pragma unroll
  for (int t = 0; t < 4; ++t) vb_cur[t] = load_b_frag(vbase + (size_t)t * vstep);

  for (int k0 = 0; k0 < SP; k0 += 32) {
    const int k1 = (k0 + 32 < SP) ? (k0 + 32) : k0;
    const v16bf pa_n = load_a_frag(&sc1[r * SP + k1 + hf * 8]);
    v16bf vb_nxt[4];
#pragma unroll
    for (int t = 0; t < 4; ++t)
      vb_nxt[t] = load_b_frag(vbase + (size_t)t * vstep + k1);

#pragma unroll
    for (int t = 0; t < 4; ++t) cacc[t] = wmma_bf16(pa, vb_cur[t], cacc[t]);

    pa = pa_n;
#pragma unroll
    for (int t = 0; t < 4; ++t) vb_cur[t] = vb_nxt[t];
  }

  // ---- scatter into c[b][(h*64+dh)*S + s] ----------------------------------
  float* cb = cbuf + (size_t)b * CPB;
#pragma unroll
  for (int t = 0; t < 4; ++t) {
    const int dh = t * 16 + r;
#pragma unroll
    for (int j = 0; j < 8; ++j) {
      const int s = q0 + j + hf * 8;
      if (s < SS) cb[(size_t)(h * DHD + dh) * SS + s] = cacc[t][j];
    }
  }
}

// ---------------------------------------------------------------------------
// GroupNorm over [B, H, Dh*S]; emits bf16 in the same flat layout, which
// (per the reference's reshape quirk) is directly the [MS x D] A-matrix of
// the output GEMM.
// ---------------------------------------------------------------------------
__global__ __launch_bounds__(256)
void k_gnorm(const float* __restrict__ cbuf, const float* __restrict__ gw,
             const float* __restrict__ gb, __bf16* __restrict__ cn) {
  __shared__ float red0[256];
  __shared__ float red1[256];
  const int bh = blockIdx.x;
  const int b = bh / HH, h = bh % HH;
  const int Nel = DHD * SS;                       // 36928
  const float*  src = cbuf + (size_t)b * CPB + (size_t)h * Nel;
  __bf16*       dst = cn   + (size_t)b * CPB + (size_t)h * Nel;
  const int tid = threadIdx.x;

  float s = 0.0f, ss = 0.0f;
  for (int i = tid; i < Nel; i += 256) {
    const float v = src[i];
    s += v; ss += v * v;
  }
  red0[tid] = s; red1[tid] = ss;
  __syncthreads();
  for (int off = 128; off > 0; off >>= 1) {
    if (tid < off) { red0[tid] += red0[tid + off]; red1[tid] += red1[tid + off]; }
    __syncthreads();
  }
  const float mu  = red0[0] / (float)Nel;
  const float var = red1[0] / (float)Nel - mu * mu;
  const float inv = rsqrtf(var + 1e-5f);

  for (int i = tid; i < Nel; i += 256) {
    const int ch = h * DHD + i / SS;
    const float g = (src[i] - mu) * inv;
    dst[i] = (__bf16)(g * gw[ch] + gb[ch]);
  }
}

// ---------------------------------------------------------------------------
// Host launcher
// ---------------------------------------------------------------------------
extern "C" void kernel_launch(void* const* d_in, const int* in_sizes, int n_in,
                              void* d_out, int out_size, void* d_ws, size_t ws_size,
                              hipStream_t stream) {
  (void)in_sizes; (void)n_in; (void)out_size; (void)ws_size;
  const float* x   = (const float*)d_in[0];
  const float* Wq  = (const float*)d_in[1];
  const float* bq  = (const float*)d_in[2];
  const float* Wk  = (const float*)d_in[3];
  const float* bk  = (const float*)d_in[4];
  const float* Wv  = (const float*)d_in[5];
  const float* bv  = (const float*)d_in[6];
  const float* Wo  = (const float*)d_in[7];
  const float* bo  = (const float*)d_in[8];
  const float* lam = (const float*)d_in[9];
  const float* gw  = (const float*)d_in[10];
  const float* gb  = (const float*)d_in[11];
  float* out = (float*)d_out;

  // workspace carve (256B aligned)
  char* w = (char*)d_ws;
  auto alloc = [&](size_t bytes) -> char* {
    char* p = w; w += (bytes + 255) & ~(size_t)255; return p;
  };
  __bf16* xb  = (__bf16*)alloc((size_t)MS * DD * 2);          // x bf16 [MS][768]
  __bf16* wqT = (__bf16*)alloc((size_t)2 * DD * DD * 2);      // [1536][768]
  __bf16* wkT = (__bf16*)alloc((size_t)2 * DD * DD * 2);      // [1536][768]
  __bf16* wvT = (__bf16*)alloc((size_t)DD * DD * 2);          // [768][768]
  __bf16* woT = (__bf16*)alloc((size_t)DD * DD * 2);          // [768][768]
  __bf16* qcb = (__bf16*)alloc((size_t)MS * 2 * DD * 2);      // [MS][1536]
  __bf16* kcb = (__bf16*)alloc((size_t)MS * 2 * DD * 2);      // [MS][1536]
  __bf16* vtb = (__bf16*)alloc((size_t)DD * BB * SP * 2);     // [768][16][608]
  float*  cbf = (float*) alloc((size_t)BB * CPB * 4);
  __bf16* cnb = (__bf16*)alloc((size_t)BB * CPB * 2);

  auto cvt = [&](const float* s, __bf16* d, int n) {
    k_cvt<<<dim3((n + 255) / 256), dim3(256), 0, stream>>>(s, d, n);
  };
  auto cvtT = [&](const float* s, __bf16* d, int K, int N) {
    k_cvt_t<<<dim3((K * N + 255) / 256), dim3(256), 0, stream>>>(s, d, K, N);
  };
  cvt (x,  xb,  MS * DD);
  cvtT(Wq, wqT, DD, 2 * DD);
  cvtT(Wk, wkT, DD, 2 * DD);
  cvtT(Wv, wvT, DD, DD);
  cvtT(Wo, woT, DD, DD);

  // zero padded V-transposed buffer (tail columns must be 0 for the PV GEMM)
  {
    const int nwords = (int)(((size_t)DD * BB * SP * 2) / 4);
    k_zero<<<dim3((nwords + 255) / 256), dim3(256), 0, stream>>>((u32*)vtb, nwords);
  }

  // projections: qc, kc (N=1536, bf16 out), v (N=768, transposed bf16 out)
  k_gemm_bf16<<<dim3(MS / 16, (2 * DD) / 128), dim3(32), 0, stream>>>(
      xb, wqT, bq, nullptr, qcb, nullptr, MS, DD, 2 * DD);
  k_gemm_bf16<<<dim3(MS / 16, (2 * DD) / 128), dim3(32), 0, stream>>>(
      xb, wkT, bk, nullptr, kcb, nullptr, MS, DD, 2 * DD);
  k_gemm_bf16<<<dim3(MS / 16, DD / 128), dim3(32), 0, stream>>>(
      xb, wvT, bv, nullptr, nullptr, vtb, MS, DD, DD);

  // fused differential attention -> c-buffer in GroupNorm layout
  k_attn<<<dim3((SS + 15) / 16, BB * HH), dim3(32), 0, stream>>>(
      qcb, kcb, vtb, lam, cbf);

  // GroupNorm -> bf16 context (flat layout == final GEMM A matrix)
  k_gnorm<<<dim3(BB * HH), dim3(256), 0, stream>>>(cbf, gw, gb, cnb);

  // output projection: f32 out
  k_gemm_bf16<<<dim3(MS / 16, DD / 128), dim3(32), 0, stream>>>(
      cnb, woT, bo, out, nullptr, nullptr, MS, DD, DD);
}
